// KGATConv_9036611191153
// MI455X (gfx1250) — compile-verified, matrix operand order checked by reading, not run on hardware
//
#include <hip/hip_runtime.h>
#include <hip/hip_bf16.h>
#include <math.h>

// ---------------------------------------------------------------------------
// KGAT conv for MI455X (gfx1250), wave32 + WMMA bf16.
//
// Roofline: compute is ~16 GFLOP (trivial for WMMA); the bound is HBM traffic
// on the [N,R,64] projection tensor (write + random per-edge row gathers).
// bf16 storage halves that dominant traffic (~1.3 GB -> ~0.7 GB => ~30us at
// 23.3 TB/s) and lets proj nearly fit the 192MB L2. All accumulation (WMMA
// C/D, softmax, segment sums) stays f32.
// ---------------------------------------------------------------------------

#define DF 64   // feature dim (fixed by the model)

typedef __attribute__((ext_vector_type(16))) __bf16 v16bf;
typedef __attribute__((ext_vector_type(8)))  float  v8f;
typedef __attribute__((ext_vector_type(4)))  unsigned v4u;
typedef __attribute__((ext_vector_type(4)))  float  v4f;
typedef __attribute__((ext_vector_type(2)))  float  v2f;

union FragBF {              // one WMMA A/B operand: 16 bf16 per lane = 8 VGPRs
  struct { v4u lo, hi; } q; // two contiguous 16B chunks per lane
  v16bf v;
};

#define WMMA_BF16(A, B, C) \
  __builtin_amdgcn_wmma_f32_16x16x32_bf16(false, (A), false, (B), (short)0, (C), false, false)

__device__ __forceinline__ unsigned short f2bf(float f) {
  unsigned u = __float_as_uint(f);
  unsigned r = u + 0x7FFFu + ((u >> 16) & 1u);  // round-to-nearest-even
  return (unsigned short)(r >> 16);
}
__device__ __forceinline__ float bf2f(unsigned hs) {
  return __uint_as_float(hs << 16);
}
// order-preserving map float -> uint so segment-max uses global_atomic_max_u32
__device__ __forceinline__ unsigned f2ord(float f) {
  unsigned u = __float_as_uint(f);
  return (u & 0x80000000u) ? ~u : (u | 0x80000000u);
}
__device__ __forceinline__ float ord2f(unsigned o) {
  unsigned u = (o & 0x80000000u) ? (o & 0x7FFFFFFFu) : ~o;
  return __uint_as_float(u);
}

// ---------------------------------------------------------------------------
// Prep: feat -> bf16, zero h_nb, init segmax/segsum
// ---------------------------------------------------------------------------
__global__ __launch_bounds__(256) void prep_feat_kernel(
    const float* __restrict__ feat, unsigned short* __restrict__ feat_bf,
    float* __restrict__ h_nb, unsigned* __restrict__ segmax, float* __restrict__ segsum,
    int N) {
  int i = blockIdx.x * blockDim.x + threadIdx.x;
  if (i < N * DF) {
    feat_bf[i] = f2bf(feat[i]);
    h_nb[i] = 0.0f;
  }
  if (i < N) {
    segmax[i] = 0x007FFFFFu;  // f2ord(-inf)
    segsum[i] = 0.0f;
  }
}

// Prep: relation_weight [r][d][o] -> bf16 transposed [r][o][d]; W1/W2 -> bf16
// (W1/W2 already have the [o][k] layout needed because out = X @ W.T)
__global__ __launch_bounds__(256) void prep_weights_kernel(
    const float* __restrict__ wrel, const float* __restrict__ W1, const float* __restrict__ W2,
    unsigned short* __restrict__ wrelT, unsigned short* __restrict__ w1b,
    unsigned short* __restrict__ w2b, int R) {
  int i = blockIdx.x * blockDim.x + threadIdx.x;
  if (i < R * DF * DF) {
    int r = i / (DF * DF);
    int rem = i - r * DF * DF;
    int d = rem / DF;
    int o = rem - d * DF;
    wrelT[(r * DF + o) * DF + d] = f2bf(wrel[i]);
  }
  if (i < DF * DF) {
    w1b[i] = f2bf(W1[i]);
    w2b[i] = f2bf(W2[i]);
  }
}

// ---------------------------------------------------------------------------
// proj[n][r][o] = sum_d feat[n][d] * Wrel[r][d][o]   (bf16 in, f32 acc, bf16 out)
// One block = 16 nodes; 8 waves each own 16x16 output tiles over (rel, colTile).
// A/B fragments: lane holds row/col (lane&15); its 16 bf16 K-values are two
// contiguous 16B chunks at K0=(lane>>4)*8 and K0+16  (ISA 7.12.2 layout).
// ---------------------------------------------------------------------------
__global__ __launch_bounds__(256) void proj_kernel(
    const unsigned short* __restrict__ feat_bf, const unsigned short* __restrict__ wrelT,
    unsigned short* __restrict__ proj_bf, int N, int R) {
  int node0 = blockIdx.x * 16;
  int wave = threadIdx.x >> 5;
  int lane = threadIdx.x & 31;
  int m  = lane & 15;
  int kb = (lane >> 4) * 8;
  bool full = (node0 + 16 <= N);         // wave-uniform: tile fully in range?

  int arow = node0 + m; if (arow >= N) arow = N - 1;
  const unsigned short* fr = feat_bf + (size_t)arow * DF + kb;
  FragBF a0, a1;                         // K = 0..31 and 32..63
  a0.q.lo = *(const v4u*)(fr);           // K kb .. kb+7
  a0.q.hi = *(const v4u*)(fr + 16);      // K kb+16 .. kb+23
  a1.q.lo = *(const v4u*)(fr + 32);
  a1.q.hi = *(const v4u*)(fr + 48);

  for (int t = wave; t < 4 * R; t += 8) {
    int rel = t >> 2;
    int ct  = t & 3;                     // 16-wide column tile of D=64
    int col = ct * 16 + m;
    const unsigned short* wr = wrelT + ((size_t)rel * DF + col) * DF + kb;  // [r][o][d]
    FragBF b0, b1;
    b0.q.lo = *(const v4u*)(wr);
    b0.q.hi = *(const v4u*)(wr + 16);
    b1.q.lo = *(const v4u*)(wr + 32);
    b1.q.hi = *(const v4u*)(wr + 48);

    v8f acc = {};
    acc = WMMA_BF16(a0.v, b0.v, acc);
    acc = WMMA_BF16(a1.v, b1.v, acc);

    // D layout: VGPR r -> row (lane>>4)*8 + r, col = ct*16 + (lane&15)
    int mbase = (lane >> 4) * 8;
    int outcol = ct * 16 + m;
    unsigned short* pb =
        proj_bf + ((size_t)(node0 + mbase) * R + rel) * DF + outcol;
    const size_t rowstride = (size_t)R * DF;
    if (full) {                          // uniform branch: straight-line stores
#pragma unroll
      for (int r8 = 0; r8 < 8; ++r8)
        pb[(size_t)r8 * rowstride] = f2bf(acc[r8]);
    } else {                             // partial tail tile only
#pragma unroll
      for (int r8 = 0; r8 < 8; ++r8)
        if (node0 + mbase + r8 < N)
          pb[(size_t)r8 * rowstride] = f2bf(acc[r8]);
    }
  }
}

// ---------------------------------------------------------------------------
// Per-edge attention logit + segment max. One wave per edge: 32 lanes x 2 elems.
// att[e] = sum_d t_r[d] * tanh(h_r[d] + embed[et][d]);  atomicMax into segmax[dst]
// ---------------------------------------------------------------------------
__global__ __launch_bounds__(256) void edge_att_kernel(
    const unsigned short* __restrict__ proj_bf, const float* __restrict__ rel_embed,
    const int* __restrict__ src, const int* __restrict__ dst, const int* __restrict__ etype,
    float* __restrict__ att, unsigned* __restrict__ segmax, int E, int R) {
  int e = blockIdx.x * (blockDim.x >> 5) + (threadIdx.x >> 5);
  int lane = threadIdx.x & 31;
  if (e >= E) return;
  int s = src[e], d = dst[e], r = etype[e];

  const unsigned short* tp = proj_bf + ((size_t)s * R + r) * DF + lane * 2;
  const unsigned short* hp = proj_bf + ((size_t)d * R + r) * DF + lane * 2;
  unsigned tu = *(const unsigned*)tp;    // 2 bf16
  unsigned hu = *(const unsigned*)hp;
  v2f em = *(const v2f*)(rel_embed + (size_t)r * DF + lane * 2);

  float t0 = bf2f(tu & 0xFFFFu), t1 = bf2f(tu >> 16);
  float h0 = bf2f(hu & 0xFFFFu), h1 = bf2f(hu >> 16);
  float v = t0 * tanhf(h0 + em.x) + t1 * tanhf(h1 + em.y);
#pragma unroll
  for (int o = 16; o > 0; o >>= 1) v += __shfl_xor(v, o, 32);
  if (lane == 0) {
    att[e] = v;
    atomicMax(segmax + d, f2ord(v));     // global_atomic_max_u32
  }
}

// ---------------------------------------------------------------------------
// e = exp(att - segmax[dst]); segsum[dst] += e  (one thread per edge)
// ---------------------------------------------------------------------------
__global__ __launch_bounds__(256) void edge_exp_kernel(
    const float* __restrict__ att, const int* __restrict__ dst,
    const unsigned* __restrict__ segmax, float* __restrict__ ev,
    float* __restrict__ segsum, int E) {
  int e = blockIdx.x * blockDim.x + threadIdx.x;
  if (e >= E) return;
  int d = dst[e];
  float x = expf(att[e] - ord2f(segmax[d]));
  ev[e] = x;
  atomicAdd(segsum + d, x);              // global_atomic_add_f32
}

// ---------------------------------------------------------------------------
// h_nb[dst] += feat[src] * (ev/segsum[dst]).  Wave per edge, 2 f32 per lane.
// feat (25.6MB) is L2-resident; atomics land in the 25.6MB h_nb in L2.
// ---------------------------------------------------------------------------
__global__ __launch_bounds__(256) void edge_agg_kernel(
    const float* __restrict__ feat, const int* __restrict__ src, const int* __restrict__ dst,
    const float* __restrict__ ev, const float* __restrict__ segsum,
    float* __restrict__ h_nb, int E) {
  int e = blockIdx.x * (blockDim.x >> 5) + (threadIdx.x >> 5);
  int lane = threadIdx.x & 31;
  if (e >= E) return;
  int s = src[e], d = dst[e];
  float a = ev[e] / segsum[d];
  v2f f = *(const v2f*)(feat + (size_t)s * DF + lane * 2);
  float* hb = h_nb + (size_t)d * DF + lane * 2;
  atomicAdd(hb + 0, f.x * a);
  atomicAdd(hb + 1, f.y * a);
}

// ---------------------------------------------------------------------------
// out = lrelu((feat + h_nb) @ W1.T) + lrelu((feat * h_nb) @ W2.T)
// Block = 16 nodes, 4 waves = 4 column tiles. A operands built on the fly
// (f32 load -> bf16 pack), B = W1b/W2b [o][k] row-major => contiguous frags.
// ---------------------------------------------------------------------------
__device__ __forceinline__ v4u pack_bf8(v4f a, v4f b) {
  v4u u;
  u.x = (unsigned)f2bf(a.x) | ((unsigned)f2bf(a.y) << 16);
  u.y = (unsigned)f2bf(a.z) | ((unsigned)f2bf(a.w) << 16);
  u.z = (unsigned)f2bf(b.x) | ((unsigned)f2bf(b.y) << 16);
  u.w = (unsigned)f2bf(b.z) | ((unsigned)f2bf(b.w) << 16);
  return u;
}

__global__ __launch_bounds__(128) void out_kernel(
    const float* __restrict__ feat, const float* __restrict__ h_nb,
    const unsigned short* __restrict__ w1b, const unsigned short* __restrict__ w2b,
    float* __restrict__ out, int N) {
  int node0 = blockIdx.x * 16;
  int ct   = threadIdx.x >> 5;           // wave = column tile 0..3
  int lane = threadIdx.x & 31;
  int m  = lane & 15;
  int kb = (lane >> 4) * 8;
  bool full = (node0 + 16 <= N);

  int arow = node0 + m; if (arow >= N) arow = N - 1;
  const float* fp = feat + (size_t)arow * DF + kb;
  const float* hp = h_nb + (size_t)arow * DF + kb;

  FragBF s0, s1, p0, p1;                 // sum = feat+h_nb, prod = feat*h_nb
#pragma unroll
  for (int c = 0; c < 4; ++c) {          // chunks at K offsets 0,16,32,48
    v4f fa = *(const v4f*)(fp + c * 16);
    v4f fb = *(const v4f*)(fp + c * 16 + 4);
    v4f ha = *(const v4f*)(hp + c * 16);
    v4f hb = *(const v4f*)(hp + c * 16 + 4);
    v4u su = pack_bf8(fa + ha, fb + hb);
    v4u pu = pack_bf8(fa * ha, fb * hb);
    if (c == 0)      { s0.q.lo = su; p0.q.lo = pu; }
    else if (c == 1) { s0.q.hi = su; p0.q.hi = pu; }
    else if (c == 2) { s1.q.lo = su; p1.q.lo = pu; }
    else             { s1.q.hi = su; p1.q.hi = pu; }
  }

  int col = ct * 16 + m;
  const unsigned short* w1p = w1b + (size_t)col * DF + kb;  // [o][k]
  const unsigned short* w2p = w2b + (size_t)col * DF + kb;
  FragBF b10, b11, b20, b21;
  b10.q.lo = *(const v4u*)(w1p);      b10.q.hi = *(const v4u*)(w1p + 16);
  b11.q.lo = *(const v4u*)(w1p + 32); b11.q.hi = *(const v4u*)(w1p + 48);
  b20.q.lo = *(const v4u*)(w2p);      b20.q.hi = *(const v4u*)(w2p + 16);
  b21.q.lo = *(const v4u*)(w2p + 32); b21.q.hi = *(const v4u*)(w2p + 48);

  v8f acc1 = {}, acc2 = {};
  acc1 = WMMA_BF16(s0.v, b10.v, acc1);
  acc1 = WMMA_BF16(s1.v, b11.v, acc1);
  acc2 = WMMA_BF16(p0.v, b20.v, acc2);
  acc2 = WMMA_BF16(p1.v, b21.v, acc2);

  int mbase = (lane >> 4) * 8;
  int outcol = ct * 16 + m;
  float* op = out + (size_t)(node0 + mbase) * DF + outcol;
  if (full) {                            // uniform branch: straight-line stores
#pragma unroll
    for (int r8 = 0; r8 < 8; ++r8) {
      float v1 = acc1[r8]; v1 = (v1 > 0.0f) ? v1 : 0.01f * v1;
      float v2 = acc2[r8]; v2 = (v2 > 0.0f) ? v2 : 0.01f * v2;
      op[(size_t)r8 * DF] = v1 + v2;
    }
  } else {
#pragma unroll
    for (int r8 = 0; r8 < 8; ++r8) {
      if (node0 + mbase + r8 < N) {
        float v1 = acc1[r8]; v1 = (v1 > 0.0f) ? v1 : 0.01f * v1;
        float v2 = acc2[r8]; v2 = (v2 > 0.0f) ? v2 : 0.01f * v2;
        op[(size_t)r8 * DF] = v1 + v2;
      }
    }
  }
}

// ---------------------------------------------------------------------------
extern "C" void kernel_launch(void* const* d_in, const int* in_sizes, int n_in,
                              void* d_out, int out_size, void* d_ws, size_t ws_size,
                              hipStream_t stream) {
  const float* feat = (const float*)d_in[0];
  const float* wrel = (const float*)d_in[1];
  const float* remb = (const float*)d_in[2];
  const float* W1   = (const float*)d_in[3];
  const float* W2   = (const float*)d_in[4];
  const int* src    = (const int*)d_in[5];
  const int* dst    = (const int*)d_in[6];
  const int* ety    = (const int*)d_in[7];
  float* out        = (float*)d_out;

  const int N = in_sizes[0] / DF;
  const int R = in_sizes[2] / DF;
  const int E = in_sizes[5];

  char* w = (char*)d_ws;
  size_t off = 0;
  auto take = [&](size_t bytes) -> char* {
    char* p = w + off;
    off = (off + bytes + 255) & ~(size_t)255;
    return p;
  };
  unsigned short* feat_bf = (unsigned short*)take((size_t)N * DF * 2);       // 12.8 MB
  unsigned short* wrelT   = (unsigned short*)take((size_t)R * DF * DF * 2);  // 128 KB
  unsigned short* w1b     = (unsigned short*)take(DF * DF * 2);
  unsigned short* w2b     = (unsigned short*)take(DF * DF * 2);
  unsigned short* projb   = (unsigned short*)take((size_t)N * R * DF * 2);   // 204.8 MB
  float*    att  = (float*)take((size_t)E * 4);                              // 6.4 MB
  float*    ev   = (float*)take((size_t)E * 4);                              // 6.4 MB
  unsigned* smax = (unsigned*)take((size_t)N * 4);
  float*    ssum = (float*)take((size_t)N * 4);
  float*    hnb  = (float*)take((size_t)N * DF * 4);                         // 25.6 MB

  const int nb16 = (N + 15) / 16;

  prep_feat_kernel<<<(N * DF + 255) / 256, 256, 0, stream>>>(feat, feat_bf, hnb, smax, ssum, N);
  prep_weights_kernel<<<(R * DF * DF + 255) / 256, 256, 0, stream>>>(wrel, W1, W2, wrelT, w1b, w2b, R);
  proj_kernel<<<nb16, 256, 0, stream>>>(feat_bf, wrelT, projb, N, R);
  edge_att_kernel<<<(E + 7) / 8, 256, 0, stream>>>(projb, remb, src, dst, ety, att, smax, E, R);
  edge_exp_kernel<<<(E + 255) / 256, 256, 0, stream>>>(att, dst, smax, ev, ssum, E);
  edge_agg_kernel<<<(E + 7) / 8, 256, 0, stream>>>(feat, src, dst, ev, ssum, hnb, E);
  out_kernel<<<nb16, 128, 0, stream>>>(feat, hnb, w1b, w2b, out, N);
}
